// Attention_58583353917557
// MI455X (gfx1250) — matrix operand order repaired and checked by reading, not compiled, not run on hardware
//
#include <hip/hip_runtime.h>

// ---------------------------------------------------------------------------
// Attention (projections + masked softmax attention) for MI455X / gfx1250.
// bf16 WMMA (v_wmma_f32_16x16x32_bf16) with fp32 accumulation, flash-style
// online softmax, TDM (tensor_load_to_lds) K-tile staging, cooperative
// shared V^T staging, explicit s_wait_dscnt / s_wait_tensorcnt.
// ---------------------------------------------------------------------------

typedef unsigned short ushort_t;
typedef __bf16 bf16_t;
typedef bf16_t bf16x8 __attribute__((ext_vector_type(8)));
typedef bf16_t v16bf  __attribute__((ext_vector_type(16)));
typedef float  v8f    __attribute__((ext_vector_type(8)));
typedef unsigned int v4u __attribute__((ext_vector_type(4)));
typedef int v4i __attribute__((ext_vector_type(4)));
typedef int v8i __attribute__((ext_vector_type(8)));

#define B_    4
#define H_    16
#define T_    2048
#define D_    128
#define ROWS_ (B_ * H_ * T_)   // 131072 flattened (b,h,t) rows

__device__ __forceinline__ v8f v8f_zero() {
    v8f z;
#pragma unroll
    for (int j = 0; j < 8; ++j) z[j] = 0.0f;
    return z;
}

__device__ __forceinline__ ushort_t f2bf_bits(float x) {
    bf16_t h = (bf16_t)x;
    return __builtin_bit_cast(ushort_t, h);
}

// A-fragment (16x32, M x K): lane half h=lane>>4 holds K = h*16 .. h*16+15.
// p points at &row[chunk*32 + h*16] of a row-major bf16 row.
__device__ __forceinline__ v16bf fragA_bf16(const ushort_t* p) {
    bf16x8 lo = *(const bf16x8*)(p);
    bf16x8 hi = *(const bf16x8*)(p + 8);
    v16bf f;
#pragma unroll
    for (int j = 0; j < 8; ++j) { f[j] = lo[j]; f[j + 8] = hi[j]; }
    return f;
}

// B-fragment (32x16, K x N): lane (n = lane&15) holds K = h*16 .. h*16+15 of column n.
// Since B[k][n] = SRC[n][k] for our row-major sources, lane n reads its row contiguously.
// p points at &row[chunk*32 + h*16].
__device__ __forceinline__ v16bf fragB_bf16(const ushort_t* p) {
    bf16x8 lo = *(const bf16x8*)(p);
    bf16x8 hi = *(const bf16x8*)(p + 8);
    v16bf f;
#pragma unroll
    for (int j = 0; j < 8; ++j) { f[j] = lo[j]; f[j + 8] = hi[j]; }
    return f;
}

// Same fragments but sourced from fp32 rows (projection kernel inputs).
__device__ __forceinline__ v16bf fragA_f32(const float* p) {
    v16bf f;
#pragma unroll
    for (int j = 0; j < 8; ++j) { f[j] = (bf16_t)p[j]; f[j + 8] = (bf16_t)p[8 + j]; }
    return f;
}
__device__ __forceinline__ v16bf fragB_f32(const float* p) {
    v16bf f;
#pragma unroll
    for (int j = 0; j < 16; ++j) f[j] = (bf16_t)p[j];
    return f;
}

// ---------------------------------------------------------------------------
// Kernel 1: Y = X @ W^T + b for q/k/v (selected by blockIdx.y), bf16 output
// into workspace. One wave per 16-row tile; 4 waves per block.
// ---------------------------------------------------------------------------
__global__ __launch_bounds__(128) void proj_kernel(
    const float* __restrict__ q, const float* __restrict__ k, const float* __restrict__ v,
    const float* __restrict__ Wq, const float* __restrict__ bq,
    const float* __restrict__ Wk, const float* __restrict__ bk,
    const float* __restrict__ Wv, const float* __restrict__ bv,
    ushort_t* __restrict__ ws)
{
    const int which = blockIdx.y;
    const float* X    = (which == 0) ? q  : (which == 1) ? k  : v;
    const float* W    = (which == 0) ? Wq : (which == 1) ? Wk : Wv;
    const float* bias = (which == 0) ? bq : (which == 1) ? bk : bv;
    ushort_t* Y = ws + (size_t)which * (size_t)ROWS_ * D_;

    const int wave = threadIdx.x >> 5;
    const int lane = threadIdx.x & 31;
    const int r = lane & 15;
    const int h = lane >> 4;
    const long row0 = ((long)blockIdx.x * 4 + wave) * 16;

    // A fragments: 16 rows of X, K = 128 in 4 chunks of 32.
    v16bf a[4];
    const float* arow = X + (row0 + r) * (long)D_;
#pragma unroll
    for (int c = 0; c < 4; ++c) a[c] = fragA_f32(arow + c * 32 + h * 16);

#pragma unroll
    for (int nt = 0; nt < 8; ++nt) {
        v8f acc = v8f_zero();
        // B[k=d][n=e] = W[e][d]: lane reads row e = nt*16 + (lane&15) of W.
        const float* wrow = W + (nt * 16 + r) * (long)D_;
#pragma unroll
        for (int c = 0; c < 4; ++c) {
            v16bf bf = fragB_f32(wrow + c * 32 + h * 16);
            acc = __builtin_amdgcn_wmma_f32_16x16x32_bf16(
                false, a[c], false, bf, (short)0, acc, false, false);
        }
        const float bb = bias[nt * 16 + r];
        // C layout: VGPR i -> row i (lanes 0-15) / i+8 (lanes 16-31), col = lane&15.
#pragma unroll
        for (int i = 0; i < 8; ++i) {
            float y = acc[i] + bb;
            Y[(size_t)(row0 + i + 8 * h) * D_ + nt * 16 + r] = f2bf_bits(y);
        }
    }
}

// ---------------------------------------------------------------------------
// Kernel 2: flash attention over projected bf16 qp/kp/vp in workspace.
// 4 waves/block share one (b,h); each wave owns 16 Q rows. Per 32-key block:
//   - K tile (32x128 bf16) DMA'd to LDS by the Tensor Data Mover (wave 0)
//   - V^T tile staged cooperatively by all 128 threads
//   - S = Q K^T, online softmax, O += P V   (all bf16 WMMA, fp32 accum)
// ---------------------------------------------------------------------------
__global__ __launch_bounds__(128) void attn_kernel(
    const ushort_t* __restrict__ ws, const int* __restrict__ mask,
    float* __restrict__ out)
{
    const ushort_t* qp = ws;
    const ushort_t* kp = ws + (size_t)ROWS_ * D_;
    const ushort_t* vp = ws + 2 * (size_t)ROWS_ * D_;

    __shared__ __align__(16) ushort_t kt[32 * 128];      // K tile, row-major (TDM dest)
    __shared__ __align__(16) ushort_t vt[128 * 32];      // vt[d*32+kk] = V[k0+kk][d]
    __shared__ __align__(16) ushort_t pt[4][16 * 32];    // per-wave P tile

    const int tid  = threadIdx.x;
    const int wave = tid >> 5;
    const int lane = tid & 31;
    const int r = lane & 15;
    const int h = lane >> 4;

    const long gt = (long)blockIdx.x * 4 + wave;  // global 16-row Q tile
    const long bh = gt >> 7;                      // same for all 4 waves (128 % 4 == 0)
    const int  q0 = (int)(gt & 127) * 16;

    const ushort_t* qb  = qp + bh * (long)T_ * D_;
    const ushort_t* vpb = vp + bh * (long)T_ * D_;
    const unsigned long long kbase =
        (unsigned long long)(const void*)(kp + bh * (long)T_ * D_);
    const unsigned lds_kt = (unsigned)(uintptr_t)(void*)&kt[0];  // LDS byte offset

    // Q fragments (16 x 128).
    v16bf qa[4];
    const ushort_t* qrow = qb + (size_t)(q0 + r) * D_;
#pragma unroll
    for (int c = 0; c < 4; ++c) qa[c] = fragA_bf16(qrow + c * 32 + h * 16);

    v8f o[8];
    float m_r[8], l_r[8];
#pragma unroll
    for (int i = 0; i < 8; ++i) { o[i] = v8f_zero(); m_r[i] = -1e30f; l_r[i] = 0.0f; }

    const float scale = 0.088388347648318447f;  // 128^-0.5

    for (int kb = 0; kb < T_ / 32; ++kb) {
        const int k0 = kb * 32;

        // ---- TDM: DMA K tile (32 rows x 128 bf16) into LDS (wave 0 only) ----
        if (wave == 0) {
            const unsigned long long ga = kbase + (unsigned long long)k0 * (D_ * 2);
            v4u g0;
            g0[0] = 1u;                                  // count=1 (valid descriptor)
            g0[1] = lds_kt;                              // lds_addr (bytes)
            g0[2] = (unsigned)(ga & 0xffffffffu);        // global_addr[31:0]
            g0[3] = (unsigned)((ga >> 32) & 0x1ffffffu)  // global_addr[56:32]
                    | 0x80000000u;                       // type=2 ("image") @ bits 127:126
            v8i g1;
            g1[0] = 0x00010000;                          // data_size=1 (2 bytes), mask=0
            g1[1] = (int)(128u << 16);                   // tensor_dim0 = 128 (low 16)
            g1[2] = (int)((unsigned)T_ << 16);           // tensor_dim1 = 2048 (low 16)
            g1[3] = (int)(128u << 16);                   // tile_dim0 = 128
            g1[4] = 32;                                  // tile_dim1 = 32, tile_dim2 = 0
            g1[5] = 128;                                 // tensor_dim0_stride = 128
            g1[6] = 0;
            g1[7] = 0;
            v4i gz; gz[0] = 0; gz[1] = 0; gz[2] = 0; gz[3] = 0;
#if __clang_major__ >= 23
            v8i gz8;
#pragma unroll
            for (int j = 0; j < 8; ++j) gz8[j] = 0;
            __builtin_amdgcn_tensor_load_to_lds(g0, g1, gz, gz, gz8, 0);
#else
            __builtin_amdgcn_tensor_load_to_lds(g0, g1, gz, gz, 0);
#endif
            __builtin_amdgcn_s_wait_tensorcnt(0);
        }

        // ---- Cooperative V^T staging: 128 threads, 4 dims x 8 keys each ----
        {
            const int d0  = (tid & 31) * 4;       // dim quad
            const int kkb = (tid >> 5) * 8;       // 8-key band
            unsigned* vt32 = (unsigned*)vt;       // index = d*16 + kk/2
#pragma unroll
            for (int j = 0; j < 4; ++j) {
                const int kk = kkb + j * 2;
                const ushort_t* vr0 = vpb + (size_t)(k0 + kk) * D_ + d0;
                uint2 va = *(const uint2*)vr0;          // V[kk  ][d0..d0+3]
                uint2 vb2 = *(const uint2*)(vr0 + D_);  // V[kk+1][d0..d0+3]
                const int kkh = kk >> 1;
                vt32[(d0 + 0) * 16 + kkh] = (va.x & 0xffffu) | (vb2.x << 16);
                vt32[(d0 + 1) * 16 + kkh] = (va.x >> 16) | (vb2.x & 0xffff0000u);
                vt32[(d0 + 2) * 16 + kkh] = (va.y & 0xffffu) | (vb2.y << 16);
                vt32[(d0 + 3) * 16 + kkh] = (va.y >> 16) | (vb2.y & 0xffff0000u);
            }
        }
        __syncthreads();  // K (TDM) + V^T staged and visible

        // ---- S = Q @ K^T (two 16x16 tiles, K rows read from LDS) ----
        v8f s01[2];
#pragma unroll
        for (int nh = 0; nh < 2; ++nh) {
            v8f s = v8f_zero();
            const ushort_t* krow = kt + (size_t)(nh * 16 + r) * D_;
#pragma unroll
            for (int c = 0; c < 4; ++c) {
                v16bf kf = fragB_bf16(krow + c * 32 + h * 16);
                s = __builtin_amdgcn_wmma_f32_16x16x32_bf16(
                    false, qa[c], false, kf, (short)0, s, false, false);
            }
            s01[nh] = s;
        }

        // ---- scale, mask (value -1e-9, faithful to reference), online softmax ----
        float alpha[8];
        ushort_t* ptw = pt[wave];
#pragma unroll
        for (int i = 0; i < 8; ++i) {
            const long qrow_g = (long)(q0 + i + 8 * h);
            const long mi = qrow_g * T_ + k0 + r;
            float x0 = s01[0][i] * scale;
            float x1 = s01[1][i] * scale;
            if (mask[mi] == 1)      x0 = -1e-9f;
            if (mask[mi + 16] == 1) x1 = -1e-9f;
            float mm = fmaxf(x0, x1);
#pragma unroll
            for (int d = 1; d <= 8; d <<= 1) mm = fmaxf(mm, __shfl_xor(mm, d, 32));
            const float m_new = fmaxf(m_r[i], mm);
            alpha[i] = __expf(m_r[i] - m_new);
            m_r[i] = m_new;
            const float p0 = __expf(x0 - m_new);
            const float p1 = __expf(x1 - m_new);
            float rs = p0 + p1;
#pragma unroll
            for (int d = 1; d <= 8; d <<= 1) rs += __shfl_xor(rs, d, 32);
            l_r[i] = l_r[i] * alpha[i] + rs;
            // P (C layout) -> LDS row-major bf16
            ptw[(i + 8 * h) * 32 + r]      = f2bf_bits(p0);
            ptw[(i + 8 * h) * 32 + 16 + r] = f2bf_bits(p1);
        }

        // Rescale running output.
#pragma unroll
        for (int nt = 0; nt < 8; ++nt)
#pragma unroll
            for (int i = 0; i < 8; ++i) o[nt][i] *= alpha[i];

        // Own-wave P stores must land before the A-fragment read.
        asm volatile("s_wait_dscnt 0" ::: "memory");

        // ---- O += P @ V (K = 32 keys) ----
        v16bf pf = fragA_bf16(ptw + r * 32 + h * 16);
#pragma unroll
        for (int nt = 0; nt < 8; ++nt) {
            v16bf vf = fragB_bf16(vt + (size_t)(nt * 16 + r) * 32 + h * 16);
            o[nt] = __builtin_amdgcn_wmma_f32_16x16x32_bf16(
                false, pf, false, vf, (short)0, o[nt], false, false);
        }
        __syncthreads();  // protect kt/vt from next iteration's overwrite
    }

    // ---- normalize and store fp32 ----
    float inv_l[8];
#pragma unroll
    for (int i = 0; i < 8; ++i) inv_l[i] = 1.0f / l_r[i];
#pragma unroll
    for (int nt = 0; nt < 8; ++nt)
#pragma unroll
        for (int i = 0; i < 8; ++i) {
            const size_t idx = (size_t)(bh * T_ + q0 + i + 8 * h) * D_ + nt * 16 + r;
            out[idx] = o[nt][i] * inv_l[i];
        }
}

// ---------------------------------------------------------------------------
// Host launcher. Inputs (dict order): q,k,v,mask,Wq,bq,Wk,bk,Wv,bv.
// Workspace: bf16 qp/kp/vp, 3 * 131072 * 128 * 2 bytes = ~100 MB.
// ---------------------------------------------------------------------------
extern "C" void kernel_launch(void* const* d_in, const int* in_sizes, int n_in,
                              void* d_out, int out_size, void* d_ws, size_t ws_size,
                              hipStream_t stream)
{
    (void)in_sizes; (void)n_in; (void)out_size; (void)ws_size;
    const float* q    = (const float*)d_in[0];
    const float* k    = (const float*)d_in[1];
    const float* v    = (const float*)d_in[2];
    const int*   mask = (const int*)d_in[3];
    const float* Wq   = (const float*)d_in[4];
    const float* bq   = (const float*)d_in[5];
    const float* Wk   = (const float*)d_in[6];
    const float* bk   = (const float*)d_in[7];
    const float* Wv   = (const float*)d_in[8];
    const float* bv   = (const float*)d_in[9];
    ushort_t* ws = (ushort_t*)d_ws;
    float* out = (float*)d_out;

    const int tiles = ROWS_ / 16;          // 8192 16-row tiles
    dim3 gp(tiles / 4, 3, 1);              // 4 waves/block, y = {q,k,v}
    proj_kernel<<<gp, 128, 0, stream>>>(q, k, v, Wq, bq, Wk, bk, Wv, bv, ws);

    dim3 ga(tiles / 4, 1, 1);
    attn_kernel<<<ga, 128, 0, stream>>>(ws, mask, out);
}
